// VideoModelCoord_13529146983103
// MI455X (gfx1250) — matrix operand order repaired and checked by reading, not compiled, not run on hardware
//
#include <hip/hip_runtime.h>
#include <hip/hip_bf16.h>
#include <math.h>

// ---------------- model constants ----------------
#define Vn   4
#define En   12
#define Tn   16
#define Dn   256
#define TDn  512
#define Hn   8
#define HDn  64
#define Bn   256
#define Sn   17   // tokens per sequence (V+E+1 == T+1 == 17)

typedef __attribute__((ext_vector_type(16))) __bf16 v16bf;
typedef __attribute__((ext_vector_type(8)))  float  v8f;

__constant__ int c_send[En] = {0,0,0,1,1,1,2,2,2,3,3,3};
__constant__ int c_recv[En] = {1,2,3,0,2,3,0,1,3,0,1,2};
__constant__ int c_recv_edges[Vn][3] = {{3,6,9},{0,7,10},{1,4,11},{2,5,8}};

// ---------------- helpers ----------------
__device__ __forceinline__ float block_sum(float v, float* red) {
  const int t = threadIdx.x;
  red[t] = v;
  __syncthreads();
  for (int s = blockDim.x >> 1; s > 0; s >>= 1) {
    if (t < s) red[t] += red[t + s];
    __syncthreads();
  }
  float r = red[0];
  __syncthreads();
  return r;
}

// ---------------- WMMA GEMM: C = act(A[M,K] @ W[K,N] + bias) ----------------
// One wave (32 lanes) computes a 16x64 output strip: 4 N-tiles sharing one
// A fragment, K stepped by 32 with v_wmma_f32_16x16x32_bf16 (f32 accumulate).
// 8 waves / block (same N strip, consecutive M tiles -> W rows hit in WGP$).
// All call sites have M%128==0, K%32==0, N%64==0.
template<int ACT>
__global__ void __launch_bounds__(256) gemm_wmma(const float* __restrict__ A,
                                                 const float* __restrict__ Wt,
                                                 const float* __restrict__ bias,
                                                 float* __restrict__ C,
                                                 int M, int N, int K) {
  const int lane   = threadIdx.x & 31;
  const int wave   = threadIdx.x >> 5;
  const int tileN4 = blockIdx.x;              // 64-wide N strip
  const int tileM  = blockIdx.y * 8 + wave;
  if (tileM * 16 >= M) return;                // wave-uniform; EXEC stays full
  const int half = lane >> 4;                 // 0: lanes 0-15, 1: lanes 16-31
  const int l    = lane & 15;
  const int col0 = tileN4 * 64 + l;
  const float* arow = A + (size_t)(tileM * 16 + l) * K;
  v8f acc0 = {}, acc1 = {}, acc2 = {}, acc3 = {};
  for (int k0 = 0; k0 < K; k0 += 32) {
    // A 16x32 bf16: lane half 0 holds K {0..7,16..23}, half 1 K {8..15,24..31}
    const float4* pa0 = (const float4*)(arow + k0 + half * 8);
    const float4* pa1 = (const float4*)(arow + k0 + 16 + half * 8);
    float4 x0 = pa0[0], x1 = pa0[1], x2 = pa1[0], x3 = pa1[1];
    v16bf a;
    a[0]=(__bf16)x0.x;  a[1]=(__bf16)x0.y;  a[2]=(__bf16)x0.z;  a[3]=(__bf16)x0.w;
    a[4]=(__bf16)x1.x;  a[5]=(__bf16)x1.y;  a[6]=(__bf16)x1.z;  a[7]=(__bf16)x1.w;
    a[8]=(__bf16)x2.x;  a[9]=(__bf16)x2.y;  a[10]=(__bf16)x2.z; a[11]=(__bf16)x2.w;
    a[12]=(__bf16)x3.x; a[13]=(__bf16)x3.y; a[14]=(__bf16)x3.z; a[15]=(__bf16)x3.w;
    // prefetch next K step (lowers to global_prefetch_b8)
    if (k0 + 32 < K) {
      __builtin_prefetch(arow + k0 + 32 + half * 8, 0, 1);
      __builtin_prefetch(Wt + (size_t)(k0 + 32 + half * 16) * N + col0, 0, 1);
    }
    // B 32x16 bf16 per tile: lane half 0 holds K {0..7,16..23}, half 1 K {8..15,24..31}
    const float* wp = Wt + (size_t)(k0 + half * 8) * N + col0;
    v16bf b;
    #pragma unroll
    for (int j = 0; j < 16; ++j) b[j] = (__bf16)wp[(size_t)(j + (j & 8)) * N];
    acc0 = __builtin_amdgcn_wmma_f32_16x16x32_bf16(false, a, false, b,
                                                   (short)0, acc0, false, false);
    #pragma unroll
    for (int j = 0; j < 16; ++j) b[j] = (__bf16)wp[(size_t)(j + (j & 8)) * N + 16];
    acc1 = __builtin_amdgcn_wmma_f32_16x16x32_bf16(false, a, false, b,
                                                   (short)0, acc1, false, false);
    #pragma unroll
    for (int j = 0; j < 16; ++j) b[j] = (__bf16)wp[(size_t)(j + (j & 8)) * N + 32];
    acc2 = __builtin_amdgcn_wmma_f32_16x16x32_bf16(false, a, false, b,
                                                   (short)0, acc2, false, false);
    #pragma unroll
    for (int j = 0; j < 16; ++j) b[j] = (__bf16)wp[(size_t)(j + (j & 8)) * N + 48];
    acc3 = __builtin_amdgcn_wmma_f32_16x16x32_bf16(false, a, false, b,
                                                   (short)0, acc3, false, false);
  }
  const float b0 = bias[col0], b1 = bias[col0 + 16];
  const float b2 = bias[col0 + 32], b3 = bias[col0 + 48];
  #pragma unroll
  for (int r = 0; r < 8; ++r) {               // C VGPR r -> row r (half 0) / r+8
    const size_t row = (size_t)(tileM * 16 + half * 8 + r);
    float y0 = acc0[r] + b0, y1 = acc1[r] + b1;
    float y2 = acc2[r] + b2, y3 = acc3[r] + b3;
    if (ACT == 1) {
      y0 = fmaxf(y0, 0.0f); y1 = fmaxf(y1, 0.0f);
      y2 = fmaxf(y2, 0.0f); y3 = fmaxf(y3, 0.0f);
    }
    if (ACT == 2) {
      y0 = 0.5f * y0 * (1.0f + erff(y0 * 0.70710678118654752f));
      y1 = 0.5f * y1 * (1.0f + erff(y1 * 0.70710678118654752f));
      y2 = 0.5f * y2 * (1.0f + erff(y2 * 0.70710678118654752f));
      y3 = 0.5f * y3 * (1.0f + erff(y3 * 0.70710678118654752f));
    }
    float* cp = C + row * N + col0;
    cp[0] = y0; cp[16] = y1; cp[32] = y2; cp[48] = y3;
  }
}

// ---------------- fused coord/flow/cat embed + LN + fusion GEMM ----------------
// one block (128 thr) per (b,t,v) row; writes bf in (B,V,T,256) layout
__global__ void __launch_bounds__(128) fuse_kernel(
    const int* __restrict__ cats, const float* __restrict__ box,
    const float* __restrict__ cat_embed,
    const float* __restrict__ cw1, const float* __restrict__ cw2,
    const float* __restrict__ fw1, const float* __restrict__ fw2,
    const float* __restrict__ ln_g, const float* __restrict__ ln_b,
    const float* __restrict__ fus_w, float* __restrict__ bf) {
  const int row = blockIdx.x;                 // (b*T + t)*V + v
  const int v = row % Vn;
  const int t = (row / Vn) % Tn;
  const int b = row / (Vn * Tn);
  const int j = threadIdx.x;                  // 0..127
  __shared__ float h[128];
  __shared__ float f[384];
  __shared__ float red[128];
  const float* bx = box + (size_t)row * 4;
  const float x0 = bx[0], x1 = bx[1], x2 = bx[2], x3 = bx[3];
  float f0, f1, f2, f3;
  if (t == 0) { f0 = f1 = f2 = f3 = 0.0f; }
  else { const float* px = bx - Vn * 4;
         f0 = x0 - px[0]; f1 = x1 - px[1]; f2 = x2 - px[2]; f3 = x3 - px[3]; }
  // coord: relu(relu(x@w1)@w2)
  h[j] = fmaxf(x0*cw1[j] + x1*cw1[128+j] + x2*cw1[256+j] + x3*cw1[384+j], 0.0f);
  __syncthreads();
  float c2 = 0.0f;
  for (int i = 0; i < 128; ++i) c2 += h[i] * cw2[i*128 + j];
  c2 = fmaxf(c2, 0.0f);
  __syncthreads();
  // flow branch
  h[j] = fmaxf(f0*fw1[j] + f1*fw1[128+j] + f2*fw1[256+j] + f3*fw1[384+j], 0.0f);
  __syncthreads();
  float fl2 = 0.0f;
  for (int i = 0; i < 128; ++i) fl2 += h[i] * fw2[i*128 + j];
  fl2 = fmaxf(fl2, 0.0f);
  const int ci = cats[row];
  f[j] = c2; f[128+j] = fl2; f[256+j] = cat_embed[(size_t)ci * 128 + j];
  __syncthreads();
  // LayerNorm over 384
  float mean = block_sum(f[j] + f[128+j] + f[256+j], red) * (1.0f/384.0f);
  float d0 = f[j]-mean, d1 = f[128+j]-mean, d2 = f[256+j]-mean;
  float var  = block_sum(d0*d0 + d1*d1 + d2*d2, red) * (1.0f/384.0f);
  float rstd = rsqrtf(var + 1e-5f);
  f[j]     = d0*rstd*ln_g[j]     + ln_b[j];
  f[128+j] = d1*rstd*ln_g[128+j] + ln_b[128+j];
  f[256+j] = d2*rstd*ln_g[256+j] + ln_b[256+j];
  __syncthreads();
  // fusion 384 -> 256 + relu, write bf (B,V,T,256)
  float* out = bf + ((size_t)(b*Vn + v)*Tn + t) * Dn;
  for (int d = j; d < Dn; d += 128) {
    float acc = 0.0f;
    for (int i = 0; i < 384; ++i) acc += f[i] * fus_w[i*Dn + d];
    out[d] = fmaxf(acc, 0.0f);
  }
}

// ---------------- node2edge gather + LayerNorm(512) ----------------
__global__ void __launch_bounds__(256) node2edge_ln(
    const float* __restrict__ src, const float* __restrict__ g,
    const float* __restrict__ be, float* __restrict__ out) {
  const int row = blockIdx.x;                 // (b*E + e)*T + t
  const int t = row % Tn;
  const int e = (row / Tn) % En;
  const int b = row / (Tn * En);
  const int tid = threadIdx.x;
  __shared__ float red[256];
  const float* s0 = src + ((size_t)(b*Vn + c_send[e])*Tn + t) * Dn;
  const float* s1 = src + ((size_t)(b*Vn + c_recv[e])*Tn + t) * Dn;
  const float v0 = s0[tid], v1 = s1[tid];
  float mean = block_sum(v0 + v1, red) * (1.0f/512.0f);
  float d0 = v0 - mean, d1 = v1 - mean;
  float rstd = rsqrtf(block_sum(d0*d0 + d1*d1, red) * (1.0f/512.0f) + 1e-5f);
  float* dst = out + (size_t)row * 512;
  dst[tid]       = d0*rstd*g[tid]     + be[tid];
  dst[256 + tid] = d1*rstd*g[256+tid] + be[256+tid];
}

// ---------------- generic row LayerNorm ----------------
__global__ void __launch_bounds__(256) ln_rows(const float* __restrict__ in,
                                               const float* __restrict__ g,
                                               const float* __restrict__ be,
                                               float* __restrict__ out, int dim) {
  const int row = blockIdx.x;
  const float* src = in + (size_t)row * dim;
  float* dst = out + (size_t)row * dim;
  __shared__ float red[256];
  const int tid = threadIdx.x;
  float s = 0.0f;
  for (int i = tid; i < dim; i += 256) s += src[i];
  float mean = block_sum(s, red) / dim;
  float vv = 0.0f;
  for (int i = tid; i < dim; i += 256) { float d = src[i]-mean; vv += d*d; }
  float rstd = rsqrtf(block_sum(vv, red) / dim + 1e-5f);
  for (int i = tid; i < dim; i += 256)
    dst[i] = (src[i]-mean)*rstd*g[i] + be[i];
}

// ---------------- edge2node scatter-mean (/3) ----------------
__global__ void __launch_bounds__(256) edge2node(const float* __restrict__ xe,
                                                 float* __restrict__ out) {
  const size_t i = (size_t)blockIdx.x * 256 + threadIdx.x;
  if (i >= (size_t)Bn*Vn*Tn*Dn) return;
  const int d = (int)(i % Dn);
  const size_t r = i / Dn;
  const int t = (int)(r % Tn);
  const size_t bv = r / Tn;
  const int v = (int)(bv % Vn);
  const int b = (int)(bv / Vn);
  float s = 0.0f;
  #pragma unroll
  for (int j = 0; j < 3; ++j) {
    const int e = c_recv_edges[v][j];
    s += xe[((size_t)(b*En + e)*Tn + t) * Dn + d];
  }
  out[i] = s * (1.0f / 3.0f);
}

// ---------------- build spatial transformer input X (b*T, 17, 512) ----------------
__global__ void __launch_bounds__(256) build_X(
    const float* __restrict__ n_out, const float* __restrict__ bf,
    const float* __restrict__ e3, const float* __restrict__ xeo,
    const float* __restrict__ sp_tok, const float* __restrict__ sp_emb,
    float* __restrict__ X) {
  const int row = blockIdx.x;                 // bt*17 + s
  const int s  = row % Sn;
  const int bt = row / Sn;
  const int t  = bt % Tn;
  const int b  = bt / Tn;
  float* dst = X + (size_t)row * TDn;
  for (int d = threadIdx.x; d < TDn; d += 256) {
    float val;
    if (s == 0) val = sp_tok[d];
    else {
      const int tok = s - 1;
      float src;
      if (tok < Vn) {
        const size_t base = ((size_t)(b*Vn + tok)*Tn + t) * Dn;
        src = (d < Dn) ? n_out[base + d] : bf[base + d - Dn];
      } else {
        const int e = tok - Vn;
        const size_t base = ((size_t)(b*En + e)*Tn + t) * Dn;
        src = (d < Dn) ? e3[base + d] : xeo[base + d - Dn];
      }
      val = src + sp_emb[tok*TDn + d];
    }
    dst[d] = val;
  }
}

// ---------------- attention: one wave per (seq, head, query) ----------------
__global__ void __launch_bounds__(256) attn_kernel(const float* __restrict__ q,
                                                   const float* __restrict__ k,
                                                   const float* __restrict__ v,
                                                   float* __restrict__ o, int R) {
  const int gw   = blockIdx.x * 8 + (threadIdx.x >> 5);
  const int lane = threadIdx.x & 31;
  const int s = gw % Sn;
  const int h = (gw / Sn) % Hn;
  const int r = gw / (Sn * Hn);
  if (r >= R) return;
  const size_t base = (size_t)r * Sn * TDn + h * HDn;   // feature index h*64 + hd
  const float q0 = q[base + (size_t)s*TDn + lane];
  const float q1 = q[base + (size_t)s*TDn + lane + 32];
  float sc[Sn];
  #pragma unroll
  for (int j = 0; j < Sn; ++j) {
    float p = q0 * k[base + (size_t)j*TDn + lane]
            + q1 * k[base + (size_t)j*TDn + lane + 32];
    #pragma unroll
    for (int m = 16; m >= 1; m >>= 1) p += __shfl_xor(p, m, 32);
    sc[j] = p * 0.125f;                       // 1/sqrt(64)
  }
  float mx = sc[0];
  #pragma unroll
  for (int j = 1; j < Sn; ++j) mx = fmaxf(mx, sc[j]);
  float sum = 0.0f;
  #pragma unroll
  for (int j = 0; j < Sn; ++j) { sc[j] = expf(sc[j] - mx); sum += sc[j]; }
  const float inv = 1.0f / sum;
  float o0 = 0.0f, o1 = 0.0f;
  #pragma unroll
  for (int j = 0; j < Sn; ++j) {
    const float aw = sc[j] * inv;
    o0 += aw * v[base + (size_t)j*TDn + lane];
    o1 += aw * v[base + (size_t)j*TDn + lane + 32];
  }
  o[base + (size_t)s*TDn + lane]      = o0;
  o[base + (size_t)s*TDn + lane + 32] = o1;
}

// ---------------- residual add + LayerNorm(512) ----------------
__global__ void __launch_bounds__(256) add_ln(const float* __restrict__ a,
                                              const float* __restrict__ b2,
                                              const float* __restrict__ g,
                                              const float* __restrict__ be,
                                              float* __restrict__ out) {
  const int row = blockIdx.x;
  const int tid = threadIdx.x;
  __shared__ float red[256];
  const float* pa = a  + (size_t)row * TDn;
  const float* pb = b2 + (size_t)row * TDn;
  const float v0 = pa[tid] + pb[tid];
  const float v1 = pa[tid+256] + pb[tid+256];
  float mean = block_sum(v0 + v1, red) * (1.0f/512.0f);
  float d0 = v0 - mean, d1 = v1 - mean;
  float rstd = rsqrtf(block_sum(d0*d0 + d1*d1, red) * (1.0f/512.0f) + 1e-5f);
  float* dst = out + (size_t)row * TDn;
  dst[tid]     = d0*rstd*g[tid]     + be[tid];
  dst[tid+256] = d1*rstd*g[tid+256] + be[tid+256];
}

// ---------------- add + LN + LN fused (layer ln2 then final norm) ----------------
__global__ void __launch_bounds__(256) add_ln_ln(const float* __restrict__ a,
                                                 const float* __restrict__ b2,
                                                 const float* __restrict__ g1,
                                                 const float* __restrict__ be1,
                                                 const float* __restrict__ g2,
                                                 const float* __restrict__ be2,
                                                 float* __restrict__ out) {
  const int row = blockIdx.x;
  const int tid = threadIdx.x;
  __shared__ float red[256];
  const float* pa = a  + (size_t)row * TDn;
  const float* pb = b2 + (size_t)row * TDn;
  const float v0 = pa[tid] + pb[tid];
  const float v1 = pa[tid+256] + pb[tid+256];
  float mean = block_sum(v0 + v1, red) * (1.0f/512.0f);
  float d0 = v0 - mean, d1 = v1 - mean;
  float rstd = rsqrtf(block_sum(d0*d0 + d1*d1, red) * (1.0f/512.0f) + 1e-5f);
  float y0 = d0*rstd*g1[tid]     + be1[tid];
  float y1 = d1*rstd*g1[tid+256] + be1[tid+256];
  float mean2 = block_sum(y0 + y1, red) * (1.0f/512.0f);
  float e0 = y0 - mean2, e1 = y1 - mean2;
  float rstd2 = rsqrtf(block_sum(e0*e0 + e1*e1, red) * (1.0f/512.0f) + 1e-5f);
  float* dst = out + (size_t)row * TDn;
  dst[tid]     = e0*rstd2*g2[tid]     + be2[tid];
  dst[tid+256] = e1*rstd2*g2[tid+256] + be2[tid+256];
}

// ---------------- build temporal input Y (b, 17, 512) from X4[:,0] ----------------
__global__ void __launch_bounds__(256) build_Y(const float* __restrict__ X4,
                                               const float* __restrict__ tok,
                                               const float* __restrict__ emb,
                                               float* __restrict__ Y) {
  const int row = blockIdx.x;                 // b*17 + s
  const int s = row % Sn;
  const int b = row / Sn;
  float* dst = Y + (size_t)row * TDn;
  for (int d = threadIdx.x; d < TDn; d += 256) {
    if (s == 0) dst[d] = tok[d];
    else {
      const int t = s - 1;
      dst[d] = X4[((size_t)(b*Tn + t) * Sn) * TDn + d] + emb[t*TDn + d];
    }
  }
}

// ---------------- host side ----------------
static void launch_gemm(int act, const float* A, const float* Wt, const float* bias,
                        float* C, int M, int N, int K, hipStream_t st) {
  dim3 g(N / 64, M / 128), b(256);
  if (act == 0)      gemm_wmma<0><<<g, b, 0, st>>>(A, Wt, bias, C, M, N, K);
  else if (act == 1) gemm_wmma<1><<<g, b, 0, st>>>(A, Wt, bias, C, M, N, K);
  else               gemm_wmma<2><<<g, b, 0, st>>>(A, Wt, bias, C, M, N, K);
}

extern "C" void kernel_launch(void* const* d_in, const int* in_sizes, int n_in,
                              void* d_out, int out_size, void* d_ws, size_t ws_size,
                              hipStream_t stream) {
  (void)in_sizes; (void)n_in; (void)out_size; (void)ws_size;
  #define IN(i) ((const float*)d_in[i])
  const int*   cats = (const int*)d_in[0];
  const float* box  = IN(1);

  float* Wp = (float*)d_ws;
  const size_t NX     = (size_t)Bn * Tn * Sn * TDn;   // 35,651,584
  const size_t NY     = (size_t)Bn * Sn * TDn;        //  2,228,224
  const size_t NBVTD  = (size_t)Bn * Vn * Tn * Dn;    //  4,194,304
  const size_t NBETD  = (size_t)Bn * En * Tn * Dn;    // 12,582,912
  const size_t NBET2D = 2 * NBETD;

  // transformer-phase buffers
  float* Xb    = Wp;
  float* qb    = Wp + NX;
  float* kb    = Wp + 2*NX;
  float* vb    = Wp + 3*NX;
  float* aob   = Wp + 4*NX;
  float* oproj = qb;                 // q dead after attention
  float* X2b   = kb;                 // k dead after attention
  float* h1b   = vb;                 // v+attn_o dead after o-proj; 142.6M floats
  float* ffnb  = qb;                 // oproj dead after add_ln

  // graph-phase buffers (all dead before qb is first written)
  float* bf    = Wp + NX;
  float* e_ln  = bf    + NBVTD;
  float* h_a   = e_ln  + NBET2D;
  float* h_b   = h_a   + NBETD;
  float* xeo   = h_b   + NBETD;      // x_edge_ori
  float* n_in2 = xeo   + NBETD;
  float* n_ln  = n_in2 + NBVTD;
  float* n_out = n_ln  + NBVTD;
  float* e3    = n_out + NBVTD;

  // ---- graph front-end ----
  fuse_kernel<<<Bn*Tn*Vn, 128, 0, stream>>>(cats, box, IN(2), IN(3), IN(4),
                                            IN(5), IN(6), IN(7), IN(8), IN(9), bf);
  // mlp1 on edges
  node2edge_ln<<<Bn*En*Tn, 256, 0, stream>>>(bf, IN(10), IN(11), e_ln);
  launch_gemm(1, e_ln, IN(12), IN(13), h_a, Bn*En*Tn, 256, 512, stream);
  launch_gemm(1, h_a,  IN(14), IN(15), h_b, Bn*En*Tn, 256, 256, stream);
  launch_gemm(0, h_b,  IN(16), IN(17), xeo, Bn*En*Tn, 256, 256, stream);
  // edge2node + mlp2
  edge2node<<<(int)(NBVTD/256), 256, 0, stream>>>(xeo, n_in2);
  ln_rows<<<Bn*Vn*Tn, 256, 0, stream>>>(n_in2, IN(18), IN(19), n_ln, 256);
  launch_gemm(1, n_ln, IN(20), IN(21), h_a,   Bn*Vn*Tn, 256, 256, stream);
  launch_gemm(1, h_a,  IN(22), IN(23), h_b,   Bn*Vn*Tn, 256, 256, stream);
  launch_gemm(0, h_b,  IN(24), IN(25), n_out, Bn*Vn*Tn, 256, 256, stream);
  // mlp3 on edges of mlp2 output
  node2edge_ln<<<Bn*En*Tn, 256, 0, stream>>>(n_out, IN(26), IN(27), e_ln);
  launch_gemm(1, e_ln, IN(28), IN(29), h_a, Bn*En*Tn, 256, 512, stream);
  launch_gemm(1, h_a,  IN(30), IN(31), h_b, Bn*En*Tn, 256, 256, stream);
  launch_gemm(0, h_b,  IN(32), IN(33), e3,  Bn*En*Tn, 256, 256, stream);

  // ---- spatial transformer (b*T sequences of 17 tokens) ----
  build_X<<<Bn*Tn*Sn, 256, 0, stream>>>(n_out, bf, e3, xeo, IN(35), IN(37), Xb);
  const int SP = 39;
  const int Msp = Bn * Tn * Sn;                      // 69632
  launch_gemm(0, Xb, IN(SP+0), IN(SP+1), qb, Msp, 512, 512, stream);
  launch_gemm(0, Xb, IN(SP+2), IN(SP+3), kb, Msp, 512, 512, stream);
  launch_gemm(0, Xb, IN(SP+4), IN(SP+5), vb, Msp, 512, 512, stream);
  attn_kernel<<<(Bn*Tn*Hn*Sn)/8, 256, 0, stream>>>(qb, kb, vb, aob, Bn*Tn);
  launch_gemm(0, aob, IN(SP+6), IN(SP+7), oproj, Msp, 512, 512, stream);
  add_ln<<<Msp, 256, 0, stream>>>(Xb, oproj, IN(SP+8), IN(SP+9), X2b);
  launch_gemm(2, X2b, IN(SP+10), IN(SP+11), h1b,  Msp, 2048, 512, stream);   // GELU
  launch_gemm(0, h1b, IN(SP+12), IN(SP+13), ffnb, Msp, 512, 2048, stream);
  add_ln_ln<<<Msp, 256, 0, stream>>>(X2b, ffnb, IN(SP+14), IN(SP+15),
                                     IN(55), IN(56), Xb);   // X4 = sp_norm(ln2(...))

  // ---- temporal transformer (b sequences of 17 tokens) ----
  float* Yb     = Wp + NX;
  float* yq     = Yb  + NY;
  float* yk     = yq  + NY;
  float* yv     = yk  + NY;
  float* yao    = yv  + NY;
  float* yh1    = yao + NY;          // 4352*2048 floats
  float* yoproj = yq;
  float* Y2     = yk;
  float* yffn   = yq;
  build_Y<<<Bn*Sn, 256, 0, stream>>>(Xb, IN(36), IN(38), Yb);
  const int TM = 57;
  const int Mtm = Bn * Sn;                           // 4352
  launch_gemm(0, Yb, IN(TM+0), IN(TM+1), yq, Mtm, 512, 512, stream);
  launch_gemm(0, Yb, IN(TM+2), IN(TM+3), yk, Mtm, 512, 512, stream);
  launch_gemm(0, Yb, IN(TM+4), IN(TM+5), yv, Mtm, 512, 512, stream);
  attn_kernel<<<(Bn*Hn*Sn)/8, 256, 0, stream>>>(yq, yk, yv, yao, Bn);
  launch_gemm(0, yao, IN(TM+6), IN(TM+7), yoproj, Mtm, 512, 512, stream);
  add_ln<<<Mtm, 256, 0, stream>>>(Yb, yoproj, IN(TM+8), IN(TM+9), Y2);
  launch_gemm(2, Y2,  IN(TM+10), IN(TM+11), yh1,  Mtm, 2048, 512, stream);   // GELU
  launch_gemm(0, yh1, IN(TM+12), IN(TM+13), yffn, Mtm, 512, 2048, stream);
  add_ln_ln<<<Mtm, 256, 0, stream>>>(Y2, yffn, IN(TM+14), IN(TM+15),
                                     IN(73), IN(74), (float*)d_out);
  #undef IN
}
